// ImplicitMotionAlignment_74088185856042
// MI455X (gfx1250) — matrix operand order repaired and checked by reading, not compiled, not run on hardware
//
#include <hip/hip_runtime.h>

typedef __attribute__((ext_vector_type(16))) __bf16 v16bf;
typedef __attribute__((ext_vector_type(8)))  __bf16 v8bf;
typedef __attribute__((ext_vector_type(2)))  __bf16 v2bf;
typedef __attribute__((ext_vector_type(8)))  float  v8f;

#define N_TOK 4096
#define CH    256
#define BM    64
#define BK    64
#define NWAVE 4
#define ATTN_SCALE 0.17677669529663687f   // (256/8)^-0.5

static __device__ __forceinline__ v16bf concat8(v8bf lo, v8bf hi) {
  return __builtin_shufflevector(lo, hi, 0,1,2,3,4,5,6,7,8,9,10,11,12,13,14,15);
}

// pe(t,ch) = ch<128 ? sin(x*f[ch]) : cos(x*f[ch-128]);  x=(t%64)/63*2-1
static __device__ __forceinline__ float pe2d(int t, int ch) {
  float x = (float)(t & 63) * (2.0f / 63.0f) - 1.0f;
  int fi = ch & 127;
  float freq = __expf(-(9.210340371976184f / 127.0f) * (float)fi);
  float ang = x * freq;
  return (ch < 128) ? __sinf(ang) : __cosf(ang);
}

// CDNA5 async LDS copy: memory -> LDS, tracked by ASYNCcnt, no VGPR staging.
static __device__ __forceinline__ void async_b128(unsigned lds_off, const void* base,
                                                  unsigned goff) {
  asm volatile("global_load_async_to_lds_b128 %0, %1, %2"
               :: "v"(lds_off), "v"(goff), "s"(base) : "memory");
}
static __device__ __forceinline__ void wait_async0() {
  asm volatile("s_wait_asynccnt 0" ::: "memory");
}

// (b, c, n) fp32  ->  (b, n, c) bf16, with positional embedding added.
__global__ __launch_bounds__(256) void prep_qk_kernel(const float* __restrict__ in,
                                                      __bf16* __restrict__ out) {
  __shared__ float tile[32][33];
  const int tx = threadIdx.x, ty = threadIdx.y;
  const int t0 = blockIdx.x * 32;
  const int c0 = blockIdx.y * 32;
  const int b  = blockIdx.z;
  const float* inb = in + (size_t)b * CH * N_TOK;
  __bf16* outb = out + (size_t)b * N_TOK * CH;
#pragma unroll
  for (int i = 0; i < 4; ++i) {
    int chl = ty + i * 8;
    tile[chl][tx] = inb[(size_t)(c0 + chl) * N_TOK + (t0 + tx)];
  }
  __syncthreads();
#pragma unroll
  for (int i = 0; i < 4; ++i) {
    int tl = ty + i * 8;
    int tt = t0 + tl;
    int ch = c0 + tx;
    float v = tile[tx][tl] + pe2d(tt, ch);
    outb[(size_t)tt * CH + ch] = (__bf16)v;
  }
}

// fp32 -> bf16 copy (V stays channel-major: that IS the B-layout P@V wants).
__global__ __launch_bounds__(256) void prep_v_kernel(const float* __restrict__ in,
                                                     __bf16* __restrict__ out) {
  int i = (blockIdx.x * 256 + threadIdx.x) * 2;
  float2 f = *(const float2*)(in + i);
  v2bf o;
  o[0] = (__bf16)f.x;
  o[1] = (__bf16)f.y;
  *(v2bf*)(out + i) = o;
}

__global__ __launch_bounds__(128) void flash_attn_kernel(const __bf16* __restrict__ qb,
                                                         const __bf16* __restrict__ kb,
                                                         const __bf16* __restrict__ vb,
                                                         float* __restrict__ out) {
  // Double-buffered tiles: 2*32KB (K) + 2*32KB (V) + 8KB (P) = 136KB (<320KB/WGP)
  __shared__ __attribute__((aligned(32))) __bf16 k_lds[2 * BK * CH];     // [buf][token][ch]
  __shared__ __attribute__((aligned(32))) __bf16 v_lds[2 * CH * BK];     // [buf][ch][token]
  __shared__ __attribute__((aligned(32))) __bf16 p_lds[NWAVE * 16 * BK]; // per-wave P

  const int tid    = threadIdx.x;
  const int wave   = tid >> 5;
  const int lane   = tid & 31;
  const int lane16 = lane & 15;
  const int hh     = (lane >> 4) & 1;   // which 16-lane half

  const int batch = blockIdx.y;
  const int q0 = blockIdx.x * BM + wave * 16;

  const __bf16* qb_b = qb + (size_t)batch * N_TOK * CH;
  const __bf16* kb_b = kb + (size_t)batch * N_TOK * CH;
  const __bf16* vb_b = vb + (size_t)batch * CH * N_TOK;

  // Q A-fragments: register-resident for the whole kernel (8 x v16bf = 64 VGPRs)
  v16bf aq[8];
  {
    const __bf16* qrow = qb_b + (size_t)(q0 + lane16) * CH;
#pragma unroll
    for (int kc = 0; kc < 8; ++kc) {
      const __bf16* p = qrow + kc * 32 + hh * 16;
      v8bf lo = *(const v8bf*)(p);        // K 0..7  (hi half: 16..23)
      v8bf hi = *(const v8bf*)(p + 8);    // K 8..15 (hi half: 24..31)
      aq[kc] = concat8(lo, hi);
    }
  }

  v8f o[16];           // 16 rows x 256 cols f32 accumulator (128 VGPRs)
  float mrun[8], lrun[8];
#pragma unroll
  for (int j = 0; j < 16; ++j)
#pragma unroll
    for (int r = 0; r < 8; ++r) o[j][r] = 0.0f;
#pragma unroll
  for (int r = 0; r < 8; ++r) { mrun[r] = -1e30f; lrun[r] = 0.0f; }

  __bf16* pw = p_lds + wave * (16 * BK);

  const unsigned klds_base = (unsigned)(uintptr_t)k_lds;   // LDS byte offset
  const unsigned vlds_base = (unsigned)(uintptr_t)v_lds;

  // Issue one K+V tile prefetch (async, ASYNCcnt-tracked, 32 x b128 per thread).
  auto prefetch_tile = [&](int kt2, int buf) {
    const unsigned kgo = (unsigned)(kt2 * BK * CH * 2);        // contiguous 32KB
    const unsigned klo = klds_base + (unsigned)buf * (BK * CH * 2);
#pragma unroll
    for (int i = 0; i < 16; ++i) {
      int idx = tid + i * 128;                                 // b128 chunk id
      async_b128(klo + idx * 16, kb_b, kgo + idx * 16);
    }
    const unsigned vlo = vlds_base + (unsigned)buf * (CH * BK * 2);
#pragma unroll
    for (int i = 0; i < 16; ++i) {
      int idx = tid + i * 128;
      int ch = idx >> 3, off8 = idx & 7;                       // 8 x 16B per ch row
      async_b128(vlo + idx * 16, vb_b,
                 (unsigned)(ch * (N_TOK * 2) + kt2 * (BK * 2) + off8 * 16));
    }
  };

  prefetch_tile(0, 0);

  for (int kt = 0; kt < N_TOK / BK; ++kt) {
    const int cur = kt & 1;
    wait_async0();        // my prefetch of buf[cur] has landed in LDS
    __syncthreads();      // everyone's prefetch landed; buf[cur^1] free to overwrite
    if (kt + 1 < N_TOK / BK) prefetch_tile(kt + 1, cur ^ 1);   // overlaps compute below

    const __bf16* kcur = k_lds + cur * (BK * CH);
    const __bf16* vcur = v_lds + cur * (CH * BK);

    // ---- S = Q * K^T (16 x 64), scaled ----
    v8f s[4];
#pragma unroll
    for (int nb = 0; nb < 4; ++nb) {
      v8f acc = {0, 0, 0, 0, 0, 0, 0, 0};
      const __bf16* krow = kcur + (size_t)(nb * 16 + lane16) * CH + hh * 16;
#pragma unroll
      for (int kc = 0; kc < 8; ++kc) {
        v16bf bfrag = *(const v16bf*)(krow + kc * 32);
        acc = __builtin_amdgcn_wmma_f32_16x16x32_bf16(false, aq[kc], false, bfrag,
                                                      (short)0, acc, false, false);
      }
#pragma unroll
      for (int r = 0; r < 8; ++r) s[nb][r] = acc[r] * ATTN_SCALE;
    }

    // ---- online softmax (row stats via 16-lane butterfly) ----
    float alpha[8];
#pragma unroll
    for (int r = 0; r < 8; ++r) {
      float t = fmaxf(fmaxf(s[0][r], s[1][r]), fmaxf(s[2][r], s[3][r]));
      t = fmaxf(t, __shfl_xor(t, 1, 32));
      t = fmaxf(t, __shfl_xor(t, 2, 32));
      t = fmaxf(t, __shfl_xor(t, 4, 32));
      t = fmaxf(t, __shfl_xor(t, 8, 32));
      float mn = fmaxf(mrun[r], t);
      alpha[r] = __expf(mrun[r] - mn);
      mrun[r] = mn;
    }
#pragma unroll
    for (int nb = 0; nb < 4; ++nb)
#pragma unroll
      for (int r = 0; r < 8; ++r) s[nb][r] = __expf(s[nb][r] - mrun[r]);
#pragma unroll
    for (int r = 0; r < 8; ++r) {
      float t = (s[0][r] + s[1][r]) + (s[2][r] + s[3][r]);
      t += __shfl_xor(t, 1, 32);
      t += __shfl_xor(t, 2, 32);
      t += __shfl_xor(t, 4, 32);
      t += __shfl_xor(t, 8, 32);
      lrun[r] = lrun[r] * alpha[r] + t;
    }
#pragma unroll
    for (int j = 0; j < 16; ++j)
#pragma unroll
      for (int r = 0; r < 8; ++r) o[j][r] *= alpha[r];

    // ---- P: C-layout -> A-layout via per-wave LDS round trip ----
#pragma unroll
    for (int nb = 0; nb < 4; ++nb)
#pragma unroll
      for (int r = 0; r < 8; ++r)
        pw[(size_t)(r + 8 * hh) * BK + nb * 16 + lane16] = (__bf16)s[nb][r];

    asm volatile("" ::: "memory");  // same-wave LDS ops retire in order; fence the compiler

    v16bf pa[2];
#pragma unroll
    for (int kc2 = 0; kc2 < 2; ++kc2) {
      const __bf16* pp = pw + (size_t)lane16 * BK + kc2 * 32 + hh * 16;
      v8bf lo = *(const v8bf*)(pp);
      v8bf hi = *(const v8bf*)(pp + 8);
      pa[kc2] = concat8(lo, hi);
    }

    // ---- O += P * V ----
#pragma unroll
    for (int j = 0; j < 16; ++j) {
      const __bf16* vrow = vcur + (size_t)(j * 16 + lane16) * BK + hh * 16;
#pragma unroll
      for (int kc2 = 0; kc2 < 2; ++kc2) {
        v16bf bv = *(const v16bf*)(vrow + kc2 * 32);
        o[j] = __builtin_amdgcn_wmma_f32_16x16x32_bf16(false, pa[kc2], false, bv,
                                                       (short)0, o[j], false, false);
      }
    }
  }

  // ---- normalize & store (b, n, c) fp32 ----
  float inv[8];
#pragma unroll
  for (int r = 0; r < 8; ++r) inv[r] = 1.0f / lrun[r];
  float* outp = out + (size_t)batch * N_TOK * CH + (size_t)q0 * CH;
#pragma unroll
  for (int j = 0; j < 16; ++j)
#pragma unroll
    for (int r = 0; r < 8; ++r)
      outp[(size_t)(r + 8 * hh) * CH + j * 16 + lane16] = o[j][r] * inv[r];
}

extern "C" void kernel_launch(void* const* d_in, const int* in_sizes, int n_in,
                              void* d_out, int out_size, void* d_ws, size_t ws_size,
                              hipStream_t stream) {
  const float* q = (const float*)d_in[0];
  const float* k = (const float*)d_in[1];
  const float* v = (const float*)d_in[2];
  float* out = (float*)d_out;

  const size_t tsz = (size_t)2 * N_TOK * CH;   // elements per tensor
  __bf16* qb = (__bf16*)d_ws;
  __bf16* kb = qb + tsz;
  __bf16* vb = kb + tsz;

  dim3 bqk(32, 8);
  dim3 gqk(N_TOK / 32, CH / 32, 2);
  prep_qk_kernel<<<gqk, bqk, 0, stream>>>(q, qb);
  prep_qk_kernel<<<gqk, bqk, 0, stream>>>(k, kb);
  prep_v_kernel<<<(unsigned)(tsz / 2 / 256), 256, 0, stream>>>(v, vb);

  dim3 gf(N_TOK / BM, 2);
  flash_attn_kernel<<<gf, 128, 0, stream>>>(qb, kb, vb, out);
}
